// BPNN_64931315581649
// MI455X (gfx1250) — matrix operand order repaired and checked by reading, not compiled, not run on hardware
//
#include <hip/hip_runtime.h>
#include <hip/hip_bf16.h>

#define NATOMS 60000
#define FD 128            // fingerprint dim
#define FH 512            // hidden
#define NE 3              // elements
#define NIMG 600
#define NNZV 2000000

typedef __attribute__((ext_vector_type(16))) __bf16 v16bf;
typedef __attribute__((ext_vector_type(8)))  __bf16 v8bf;
typedef __attribute__((ext_vector_type(8)))  float  v8f;
typedef __attribute__((ext_vector_type(4)))  float  v4f;

// Branchless tanh using HW transcendentals: v_exp_f32 + v_rcp_f32 (5 VALU ops).
// Saturates correctly: x>>0 -> exp=inf -> rcp=0 -> 1 ; x<<0 -> exp=0 -> 1-2 = -1.
__device__ inline float fast_tanh(float x) {
    float e = __expf(2.0f * x);
    return __builtin_fmaf(-2.0f, __builtin_amdgcn_rcpf(e + 1.0f), 1.0f);
}

// A-fragment (16x32 bf16) from row-major bf16 LDS buffer, row stride FH.
// lane<16: row M=row, elems0-7 = K kt*32..+7, elems8-15 = K kt*32+16..+23
// lane>=16: same rows, K offset +8 (per ISA 16-bit A layout).
__device__ inline v16bf frag_a_lds(const __bf16* buf, int row, int half, int kt) {
    const __bf16* p = buf + row * FH + kt * 32 + half * 8;
    v8bf lo = *(const v8bf*)(p);
    v8bf hi = *(const v8bf*)(p + 16);
    return __builtin_shufflevector(lo, hi, 0,1,2,3,4,5,6,7,8,9,10,11,12,13,14,15);
}

__device__ inline v8f wmma_bf16(v16bf a, v16bf b, v8f c) {
    return __builtin_amdgcn_wmma_f32_16x16x32_bf16(false, a, false, b, (short)0, c,
                                                   false, false);
}

// ---------------- kernel 0: zero outputs + meta ----------------
__global__ void k_zero(float* __restrict__ out, int n_out, int* __restrict__ meta) {
    int i = blockIdx.x * blockDim.x + threadIdx.x;
    if (i < n_out) out[i] = 0.0f;
    if (i < 16) meta[i] = 0;
}

// ---------------- kernel 1a: count atoms per element ----------------
__global__ void k_count(const int* __restrict__ an, int* __restrict__ meta) {
    int i = blockIdx.x * blockDim.x + threadIdx.x;
    if (i < NATOMS) atomicAdd(&meta[an[i]], 1);
}

// ---------------- kernel 1b: tiny exclusive scan (E=3) ----------------
__global__ void k_scan(int* __restrict__ meta) {
    meta[4] = 0;
    meta[5] = meta[0];
    meta[6] = meta[0] + meta[1];
    meta[8] = meta[4];
    meta[9] = meta[5];
    meta[10] = meta[6];
}

// ---------------- kernel 1c: fill permutation (counting sort) ----------------
__global__ void k_fill(const int* __restrict__ an, int* __restrict__ meta,
                       int* __restrict__ perm) {
    int i = blockIdx.x * blockDim.x + threadIdx.x;
    if (i < NATOMS) {
        int p = atomicAdd(&meta[8 + an[i]], 1);
        perm[p] = i;
    }
}

// ---------------- kernel 2: f32 -> bf16 weights (+ transposes for forward) ----------
__global__ void k_convert(const float* __restrict__ W1, const float* __restrict__ W2,
                          __bf16* __restrict__ w1bf, __bf16* __restrict__ w1tbf,
                          __bf16* __restrict__ w2bf, __bf16* __restrict__ w2tbf) {
    int i = blockIdx.x * blockDim.x + threadIdx.x;
    const int n1 = NE * FD * FH;
    const int n2 = NE * FH * FH;
    if (i < n1) {
        __bf16 b = (__bf16)W1[i];
        w1bf[i] = b;
        int e = i / (FD * FH);
        int rem = i - e * FD * FH;
        int d = rem / FH;
        int h = rem - d * FH;
        w1tbf[(size_t)e * FH * FD + (size_t)h * FD + d] = b;   // W1T[e][h][d] = W1[e][d][h]
    } else if (i < n1 + n2) {
        int j = i - n1;
        __bf16 b = (__bf16)W2[j];
        w2bf[j] = b;
        int e = j / (FH * FH);
        int rem = j - e * FH * FH;
        int a = rem / FH;
        int c = rem - a * FH;
        w2tbf[(size_t)e * FH * FH + (size_t)c * FH + a] = b;   // W2T[e][k][h] = W2[e][h][k]
    }
}

// -------- kernel 3: fused fwd+bwd MLP, one wave per 32-atom tile (2 WMMA M-tiles) --------
__global__ __launch_bounds__(32)
void k_mlp_tile(const float* __restrict__ fp,
                const int* __restrict__ image_idx,
                const float* __restrict__ b1, const float* __restrict__ b2,
                const float* __restrict__ W3, const float* __restrict__ b3,
                const __bf16* __restrict__ w1bf, const __bf16* __restrict__ w1tbf,
                const __bf16* __restrict__ w2bf, const __bf16* __restrict__ w2tbf,
                const int* __restrict__ meta, const int* __restrict__ perm,
                float* __restrict__ dEdfp, float* __restrict__ energy)
{
    __shared__ __bf16 h1[32 * FH];   // layer1 activations, later overwritten by delta1
    __shared__ __bf16 h2[32 * FH];   // layer2 activations, later overwritten by delta2

    // ---- map block -> (element, local 32-atom tile) ----
    int T = blockIdx.x;
    int e = -1, lt = 0, acc_t = 0, cnt = 0, off = 0;
    #pragma unroll
    for (int i = 0; i < NE; i++) {
        int c = meta[i];
        int t = (c + 31) >> 5;
        if (e < 0 && T < acc_t + t) { e = i; lt = T - acc_t; cnt = c; off = meta[4 + i]; }
        acc_t += t;
    }
    if (e < 0) return;
    int base  = off + lt * 32;
    int valid = min(32, cnt - lt * 32);

    int lane = threadIdx.x;
    int half = lane >> 4;
    int mrow = lane & 15;

    int aid0 = perm[base + min(mrow, valid - 1)];
    int aid1 = perm[base + min(16 + mrow, valid - 1)];

    // ---- preload fingerprint A-fragments for both M-tiles (f32 -> bf16) ----
    v16bf afp0[4], afp1[4];
    {
        const float* p0 = fp + (size_t)aid0 * FD;
        const float* p1 = fp + (size_t)aid1 * FD;
        #pragma unroll
        for (int kt = 0; kt < 4; kt++) {
            const float* q0 = p0 + kt * 32 + half * 8;
            const float* q1 = p1 + kt * 32 + half * 8;
            v4f a0 = *(const v4f*)(q0),     a1 = *(const v4f*)(q0 + 4);
            v4f a2 = *(const v4f*)(q0 + 16), a3 = *(const v4f*)(q0 + 20);
            v4f c0 = *(const v4f*)(q1),     c1 = *(const v4f*)(q1 + 4);
            v4f c2 = *(const v4f*)(q1 + 16), c3 = *(const v4f*)(q1 + 20);
            v16bf x, y;
            #pragma unroll
            for (int i = 0; i < 4; i++) {
                x[i] = (__bf16)a0[i]; x[4+i] = (__bf16)a1[i];
                x[8+i] = (__bf16)a2[i]; x[12+i] = (__bf16)a3[i];
                y[i] = (__bf16)c0[i]; y[4+i] = (__bf16)c1[i];
                y[8+i] = (__bf16)c2[i]; y[12+i] = (__bf16)c3[i];
            }
            afp0[kt] = x; afp1[kt] = y;
        }
    }

    // ---- GEMM1: h1 = tanh(X @ W1 + b1)   [32x128]x[128x512] ----
    const __bf16* W1e = w1bf + (size_t)e * FD * FH;
    for (int nt = 0; nt < FH / 16; nt++) {
        float bv = b1[(size_t)e * FH + nt * 16 + mrow];
        v8f acc0 = {bv, bv, bv, bv, bv, bv, bv, bv};
        v8f acc1 = acc0;
        v16bf bf[4];
        #pragma unroll
        for (int kt = 0; kt < 4; kt++)
        {
            const __bf16* q = w1tbf + (size_t)e * FH * FD + (size_t)(nt * 16 + mrow) * FD + kt * 32 + half * 8;
            bf[kt] = __builtin_shufflevector(*(const v8bf*)(q), *(const v8bf*)(q + 16), 0,1,2,3,4,5,6,7,8,9,10,11,12,13,14,15);
        }
        #pragma unroll
        for (int kt = 0; kt < 4; kt++) {
            acc0 = wmma_bf16(afp0[kt], bf[kt], acc0);
            acc1 = wmma_bf16(afp1[kt], bf[kt], acc1);
        }
        #pragma unroll
        for (int r = 0; r < 8; r++) {
            int m = r + 8 * half, n = nt * 16 + mrow;
            h1[m * FH + n]        = (__bf16)fast_tanh(acc0[r]);
            h1[(m + 16) * FH + n] = (__bf16)fast_tanh(acc1[r]);
        }
    }

    // ---- GEMM2: h2 = tanh(h1 @ W2 + b2)   [32x512]x[512x512] ----
    const __bf16* W2e = w2bf + (size_t)e * FH * FH;
    for (int nt = 0; nt < FH / 16; nt++) {
        float bv = b2[(size_t)e * FH + nt * 16 + mrow];
        v8f acc0 = {bv, bv, bv, bv, bv, bv, bv, bv};
        v8f acc1 = acc0;
        for (int kc = 0; kc < 4; kc++) {
            v16bf bf[4], a0[4], a1[4];
            #pragma unroll
            for (int j = 0; j < 4; j++) {
                int kt = kc * 4 + j;
                bf[j] = frag_a_lds(w2tbf + (size_t)e * FH * FH, nt * 16 + mrow, half, kt);
                a0[j] = frag_a_lds(h1, mrow, half, kt);
                a1[j] = frag_a_lds(h1, 16 + mrow, half, kt);
            }
            #pragma unroll
            for (int j = 0; j < 4; j++) {
                acc0 = wmma_bf16(a0[j], bf[j], acc0);
                acc1 = wmma_bf16(a1[j], bf[j], acc1);
            }
        }
        #pragma unroll
        for (int r = 0; r < 8; r++) {
            int m = r + 8 * half, n = nt * 16 + mrow;
            h2[m * FH + n]        = (__bf16)fast_tanh(acc0[r]);
            h2[(m + 16) * FH + n] = (__bf16)fast_tanh(acc1[r]);
        }
    }
    __syncthreads();

    // ---- per-atom energy: e = h2 . W3 + b3 (lanes cover one row cooperatively) ----
    const float* W3e = W3 + (size_t)e * FH;
    float b3v = b3[e];
    for (int m = 0; m < 32; m++) {
        const __bf16* prow = h2 + m * FH + lane * 16;    // conflict-free: lanes span banks
        const float*  wrow = W3e + lane * 16;
        float part = 0.0f;
        #pragma unroll
        for (int t2 = 0; t2 < 16; t2++) part += (float)prow[t2] * wrow[t2];
        #pragma unroll
        for (int o = 16; o > 0; o >>= 1) part += __shfl_xor(part, o, 32);
        if (lane == 0 && m < valid)
            atomicAdd(&energy[image_idx[perm[base + m]]], part + b3v);
    }

    // ---- delta2 = (1 - h2^2) * W3  (overwrite h2 in LDS) ----
    for (int idx = lane; idx < 32 * FH; idx += 32) {
        int j = idx & (FH - 1);
        float v = (float)h2[idx];
        h2[idx] = (__bf16)((1.0f - v * v) * W3e[j]);
    }
    __syncthreads();

    // ---- backward GEMM2: g1 = delta2 @ W2^T ; delta1 = (1-h1^2)*g1 (overwrite h1) ----
    const __bf16* W2Te = w2tbf + (size_t)e * FH * FH;
    for (int nt = 0; nt < FH / 16; nt++) {
        v8f acc0 = {0.f, 0.f, 0.f, 0.f, 0.f, 0.f, 0.f, 0.f};
        v8f acc1 = acc0;
        for (int kc = 0; kc < 4; kc++) {
            v16bf bf[4], a0[4], a1[4];
            #pragma unroll
            for (int j = 0; j < 4; j++) {
                int kt = kc * 4 + j;
                bf[j] = frag_a_lds(W2e, nt * 16 + mrow, half, kt);
                a0[j] = frag_a_lds(h2, mrow, half, kt);
                a1[j] = frag_a_lds(h2, 16 + mrow, half, kt);
            }
            #pragma unroll
            for (int j = 0; j < 4; j++) {
                acc0 = wmma_bf16(a0[j], bf[j], acc0);
                acc1 = wmma_bf16(a1[j], bf[j], acc1);
            }
        }
        #pragma unroll
        for (int r = 0; r < 8; r++) {
            int m = r + 8 * half, n = nt * 16 + mrow;
            float hv0 = (float)h1[m * FH + n];
            float hv1 = (float)h1[(m + 16) * FH + n];
            h1[m * FH + n]        = (__bf16)((1.0f - hv0 * hv0) * acc0[r]);
            h1[(m + 16) * FH + n] = (__bf16)((1.0f - hv1 * hv1) * acc1[r]);
        }
    }
    __syncthreads();

    // ---- backward GEMM1: dE/dfp = delta1 @ W1^T   [32x512]x[512x128] -> f32 scatter ----
    int sid0[8], sid1[8];
    #pragma unroll
    for (int r = 0; r < 8; r++) {
        int m = r + 8 * half;
        sid0[r] = (m < valid)      ? perm[base + m]      : -1;
        sid1[r] = (m + 16 < valid) ? perm[base + m + 16] : -1;
    }
    const __bf16* W1Te = w1tbf + (size_t)e * FH * FD;
    for (int nt = 0; nt < FD / 16; nt++) {
        v8f acc0 = {0.f, 0.f, 0.f, 0.f, 0.f, 0.f, 0.f, 0.f};
        v8f acc1 = acc0;
        for (int kc = 0; kc < 4; kc++) {
            v16bf bf[4], a0[4], a1[4];
            #pragma unroll
            for (int j = 0; j < 4; j++) {
                int kt = kc * 4 + j;
                bf[j] = frag_a_lds(W1e, nt * 16 + mrow, half, kt);
                a0[j] = frag_a_lds(h1, mrow, half, kt);
                a1[j] = frag_a_lds(h1, 16 + mrow, half, kt);
            }
            #pragma unroll
            for (int j = 0; j < 4; j++) {
                acc0 = wmma_bf16(a0[j], bf[j], acc0);
                acc1 = wmma_bf16(a1[j], bf[j], acc1);
            }
        }
        #pragma unroll
        for (int r = 0; r < 8; r++) {
            int col = nt * 16 + mrow;
            if (sid0[r] >= 0) dEdfp[(size_t)sid0[r] * FD + col] = acc0[r];
            if (sid1[r] >= 0) dEdfp[(size_t)sid1[r] * FD + col] = acc1[r];
        }
    }
}

// -------- kernel 4: sparse COO force scatter: forces[c] -= v * dEdfp[r] --------
__global__ void k_forces(const int* __restrict__ rows, const int* __restrict__ cols,
                         const float* __restrict__ vals, const float* __restrict__ dEdfp,
                         float* __restrict__ forces) {
    int i = blockIdx.x * blockDim.x + threadIdx.x;
    if (i < NNZV) {
        atomicAdd(&forces[cols[i]], -vals[i] * dEdfp[rows[i]]);
    }
}

static inline size_t align256(size_t x) { return (x + 255) & ~(size_t)255; }

extern "C" void kernel_launch(void* const* d_in, const int* in_sizes, int n_in,
                              void* d_out, int out_size, void* d_ws, size_t ws_size,
                              hipStream_t stream) {
    const float* fp   = (const float*)d_in[0];
    const int*   an   = (const int*)d_in[1];
    const int*   img  = (const int*)d_in[2];
    const int*   rows = (const int*)d_in[3];
    const int*   cols = (const int*)d_in[4];
    const float* vals = (const float*)d_in[5];
    const float* W1   = (const float*)d_in[6];
    const float* b1   = (const float*)d_in[7];
    const float* W2   = (const float*)d_in[8];
    const float* b2   = (const float*)d_in[9];
    const float* W3   = (const float*)d_in[10];
    const float* b3   = (const float*)d_in[11];

    float* out    = (float*)d_out;           // [600 energy][180000 forces]
    float* energy = out;
    float* forces = out + NIMG;

    char* ws = (char*)d_ws;
    size_t o_meta = 0;
    size_t o_perm = 256;
    size_t o_w1   = align256(o_perm + (size_t)NATOMS * 4);
    size_t o_w1t  = align256(o_w1  + (size_t)NE * FD * FH * 2);
    size_t o_w2   = align256(o_w1t + (size_t)NE * FH * FD * 2);
    size_t o_w2t  = align256(o_w2  + (size_t)NE * FH * FH * 2);
    size_t o_dE   = align256(o_w2t + (size_t)NE * FH * FH * 2);

    int*    meta  = (int*)(ws + o_meta);
    int*    perm  = (int*)(ws + o_perm);
    __bf16* w1bf  = (__bf16*)(ws + o_w1);
    __bf16* w1tbf = (__bf16*)(ws + o_w1t);
    __bf16* w2bf  = (__bf16*)(ws + o_w2);
    __bf16* w2tbf = (__bf16*)(ws + o_w2t);
    float*  dEdfp = (float*)(ws + o_dE);

    const int BT = 256;
    int n_out = NIMG + 3 * NATOMS;  // 180600

    k_zero<<<(n_out + BT - 1) / BT, BT, 0, stream>>>(out, n_out, meta);
    k_count<<<(NATOMS + BT - 1) / BT, BT, 0, stream>>>(an, meta);
    k_scan<<<1, 1, 0, stream>>>(meta);
    k_fill<<<(NATOMS + BT - 1) / BT, BT, 0, stream>>>(an, meta, perm);

    int n_conv = NE * FD * FH + NE * FH * FH;  // 983040
    k_convert<<<(n_conv + BT - 1) / BT, BT, 0, stream>>>(W1, W2, w1bf, w1tbf, w2bf, w2tbf);

    // max 32-atom tiles across elements: ceil(N/32) + (E-1) partial tiles
    int n_tiles = (NATOMS + 31) / 32 + (NE - 1);   // 1877
    k_mlp_tile<<<n_tiles, 32, 0, stream>>>(fp, img, b1, b2, W3, b3,
                                           w1bf, w1tbf, w2bf, w2tbf,
                                           meta, perm, dEdfp, energy);

    k_forces<<<(NNZV + BT - 1) / BT, BT, 0, stream>>>(rows, cols, vals, dEdfp, forces);
}
